// SpatioTemporalAttention_35682588295600
// MI455X (gfx1250) — compile-verified, matrix operand order checked by reading, not run on hardware
//
#include <hip/hip_runtime.h>
#include <hip/hip_bf16.h>

// MI455X / gfx1250, wave32. bf16 WMMA path: v_wmma_f32_16x16x32_bf16.
// GEMM tile staging uses async global->LDS DMA (ASYNCcnt) when available.

typedef __bf16  bf16_t;
typedef __bf16  bf16x16 __attribute__((ext_vector_type(16)));
typedef float   v8f     __attribute__((ext_vector_type(8)));
typedef int     v4i     __attribute__((ext_vector_type(4)));

#define MROWS 32768   // 8 * 4096
#define CDIM  1024
#define HEADS 16
#define LSEQ  4096

#if __has_builtin(__builtin_amdgcn_global_load_async_to_lds_b128) && \
    __has_builtin(__builtin_amdgcn_s_wait_asynccnt)
#define USE_ASYNC_LDS 1
#endif

// Copy 16 bytes global -> LDS. Async DMA path bypasses VGPR staging and is
// tracked with ASYNCcnt; fallback is a regular b128 load + ds_store.
__device__ __forceinline__ void copy16_g2lds(const bf16_t* __restrict__ g,
                                             bf16_t* __restrict__ l) {
#ifdef USE_ASYNC_LDS
    __builtin_amdgcn_global_load_async_to_lds_b128(
        (__attribute__((address_space(1))) v4i*)g,
        (__attribute__((address_space(3))) v4i*)l, 0, 0);
#else
    *(uint4*)l = *(const uint4*)g;
#endif
}

__device__ __forceinline__ void wait_g2lds() {
#ifdef USE_ASYNC_LDS
    __builtin_amdgcn_s_wait_asynccnt(0);
#endif
}

// ---------------------------------------------------------------------------
// f32 -> bf16 elementwise convert
// ---------------------------------------------------------------------------
__global__ __launch_bounds__(256) void convert_bf16(const float* __restrict__ in,
                                                    bf16_t* __restrict__ out,
                                                    size_t n) {
    size_t i = (size_t)blockIdx.x * blockDim.x + threadIdx.x;
    size_t stride = (size_t)gridDim.x * blockDim.x;
    for (; i < n; i += stride) out[i] = (bf16_t)in[i];
}

// ---------------------------------------------------------------------------
// W [rows,cols] f32 -> WT [cols,rows] bf16 (tiled transpose)
// ---------------------------------------------------------------------------
__global__ __launch_bounds__(256) void transpose_convert(const float* __restrict__ W,
                                                         bf16_t* __restrict__ WT,
                                                         int rows, int cols) {
    __shared__ float tile[32][33];
    const int c0 = blockIdx.x * 32, r0 = blockIdx.y * 32;
    const int tx = threadIdx.x, ty = threadIdx.y;  // block (32,8)
    for (int i = ty; i < 32; i += 8) {
        int r = r0 + i, c = c0 + tx;
        if (r < rows && c < cols) tile[i][tx] = W[(size_t)r * cols + c];
    }
    __syncthreads();
    for (int i = ty; i < 32; i += 8) {
        int c = c0 + i, r = r0 + tx;
        if (c < cols && r < rows) WT[(size_t)c * rows + r] = (bf16_t)tile[tx][i];
    }
}

// ---------------------------------------------------------------------------
// Main GEMM: Out[M,1024] = A[M,1024](bf16) @ B[1024,1024] + epilogue
// B supplied transposed: BT[n][k] bf16, so each lane's B-fragment is a
// contiguous row slice (same packing as A fragments per CDNA5 16-bit layout).
// Block: 256 threads = 8 waves; block tile 128(M) x 128(N);
// wave tile 32(M) x 64(N) = 2x4 WMMA accumulators; K tiled by 32, LDS
// double-buffered, next tile fetched by async DMA while WMMAs run.
// MODE: 1 = bias->bf16, 2 = (bias)*gate->bf16, 3 = bias+addend->bf16,
//       4 = bias->f32
// ---------------------------------------------------------------------------
template <int MODE>
__global__ __launch_bounds__(256) void gemm_bf16(const bf16_t* __restrict__ A,
                                                 const bf16_t* __restrict__ BT,
                                                 const float* __restrict__ bias,
                                                 const float* __restrict__ gate,
                                                 const bf16_t* __restrict__ addend,
                                                 void* __restrict__ Out) {
    __shared__ bf16_t lA[2][128 * 32];
    __shared__ bf16_t lB[2][128 * 32];

    const int tid  = threadIdx.x;
    const int wv   = tid >> 5;
    const int lane = tid & 31;
    const int half = lane >> 4;
    const int l16  = lane & 15;

    const int blockN = blockIdx.x * 128;
    const int blockM = blockIdx.y * 128;

    const int wm = (wv & 3) * 32;   // wave M offset in block tile
    const int wn = (wv >> 2) * 64;  // wave N offset in block tile

    v8f acc[2][4];
#pragma unroll
    for (int i = 0; i < 2; i++)
#pragma unroll
        for (int j = 0; j < 4; j++) acc[i][j] = (v8f){0, 0, 0, 0, 0, 0, 0, 0};

    const bf16_t* gAbase = A + (size_t)blockM * CDIM;
    const bf16_t* gBbase = BT + (size_t)blockN * CDIM;

    auto loadTiles = [&](int kt, int buf) {
        const bf16_t* gA = gAbase + kt * 32;
        const bf16_t* gB = gBbase + kt * 32;
#pragma unroll
        for (int i = 0; i < 2; i++) {
            int slot = tid + i * 256;   // 512 slots of 16B cover 128x32 bf16
            int row  = slot >> 2;
            int seg  = (slot & 3) * 8;  // bf16 elements
            copy16_g2lds(gA + (size_t)row * CDIM + seg, &lA[buf][row * 32 + seg]);
            copy16_g2lds(gB + (size_t)row * CDIM + seg, &lB[buf][row * 32 + seg]);
        }
    };

    loadTiles(0, 0);
    wait_g2lds();
    __syncthreads();

    for (int kt = 0; kt < CDIM / 32; ++kt) {
        const int buf = kt & 1;
        if (kt + 1 < CDIM / 32) loadTiles(kt + 1, buf ^ 1);

        // Build fragments (per-lane: K {0..7,16..23} for lanes 0-15,
        // K {8..15,24..31} for lanes 16-31).
        bf16x16 fa[2], fb[4];
#pragma unroll
        for (int i = 0; i < 2; i++) {
            const bf16_t* p = &lA[buf][(wm + i * 16 + l16) * 32 + half * 8];
            uint4* q = reinterpret_cast<uint4*>(&fa[i]);
            q[0] = *(const uint4*)p;
            q[1] = *(const uint4*)(p + 16);
        }
#pragma unroll
        for (int j = 0; j < 4; j++) {
            const bf16_t* p = &lB[buf][(wn + j * 16 + l16) * 32 + half * 8];
            uint4* q = reinterpret_cast<uint4*>(&fb[j]);
            q[0] = *(const uint4*)p;
            q[1] = *(const uint4*)(p + 16);
        }
#pragma unroll
        for (int i = 0; i < 2; i++)
#pragma unroll
            for (int j = 0; j < 4; j++)
                acc[i][j] = __builtin_amdgcn_wmma_f32_16x16x32_bf16(
                    false, fa[i], false, fb[j], (short)0, acc[i][j], false, false);

        wait_g2lds();
        __syncthreads();
    }

    // Epilogue. C layout: VGPR r, lanes 0-15 -> (M=r, N=lane); lanes 16-31 ->
    // (M=r+8, N=lane-16).
#pragma unroll
    for (int i = 0; i < 2; i++) {
#pragma unroll
        for (int j = 0; j < 4; j++) {
            const int col = blockN + wn + j * 16 + l16;
            const float b = bias[col];
#pragma unroll
            for (int r = 0; r < 8; r++) {
                const int row = blockM + wm + i * 16 + half * 8 + r;
                const size_t idx = (size_t)row * CDIM + col;
                float v = acc[i][j][r] + b;
                if (MODE == 1) {
                    ((bf16_t*)Out)[idx] = (bf16_t)v;
                } else if (MODE == 2) {
                    v *= gate[(size_t)(row >> 12) * CDIM + col];  // batch = row/4096
                    ((bf16_t*)Out)[idx] = (bf16_t)v;
                } else if (MODE == 3) {
                    v += (float)addend[idx];
                    ((bf16_t*)Out)[idx] = (bf16_t)v;
                } else {
                    ((float*)Out)[idx] = v;
                }
            }
        }
    }
}

// ---------------------------------------------------------------------------
// Logit GEMM: out[M,16] = scale*( (X (* gate)) @ WaT^T + ba )
// One wave per 16-row tile, one WMMA per K-step (N=16).
// gate == nullptr -> plain; else X gated per (batch, channel).
// ---------------------------------------------------------------------------
__global__ __launch_bounds__(256) void logits_kernel(const bf16_t* __restrict__ X,
                                                     const float* __restrict__ gate,
                                                     const bf16_t* __restrict__ WaT, // [16,1024]
                                                     const float* __restrict__ ba,   // [16]
                                                     float* __restrict__ out) {      // [M,16]
    const int tid = threadIdx.x;
    const int wv = tid >> 5, lane = tid & 31;
    const int half = lane >> 4, l16 = lane & 15;
    const int rowBase = (blockIdx.x * 8 + wv) * 16;
    const int batch = rowBase >> 12;

    v8f acc = (v8f){0, 0, 0, 0, 0, 0, 0, 0};
    for (int kt = 0; kt < 32; ++kt) {
        bf16x16 fa, fb;
        const bf16_t* pa = X + (size_t)(rowBase + l16) * CDIM + kt * 32 + half * 8;
        {
            uint4* q = reinterpret_cast<uint4*>(&fa);
            q[0] = *(const uint4*)pa;
            q[1] = *(const uint4*)(pa + 16);
        }
        if (gate) {
            const float* g = gate + (size_t)batch * CDIM + kt * 32 + half * 8;
            bf16_t* e = reinterpret_cast<bf16_t*>(&fa);
#pragma unroll
            for (int k = 0; k < 8; k++) e[k] = (bf16_t)((float)e[k] * g[k]);
#pragma unroll
            for (int k = 0; k < 8; k++) e[8 + k] = (bf16_t)((float)e[8 + k] * g[16 + k]);
        }
        const bf16_t* pb = WaT + (size_t)l16 * CDIM + kt * 32 + half * 8;
        {
            uint4* q = reinterpret_cast<uint4*>(&fb);
            q[0] = *(const uint4*)pb;
            q[1] = *(const uint4*)(pb + 16);
        }
        acc = __builtin_amdgcn_wmma_f32_16x16x32_bf16(false, fa, false, fb, (short)0,
                                                      acc, false, false);
    }
    const float scale = 0.125f;  // 1/sqrt(DH=64)
    const float b = ba[l16];
#pragma unroll
    for (int r = 0; r < 8; r++) {
        const int row = rowBase + half * 8 + r;
        out[(size_t)row * HEADS + l16] = (acc[r] + b) * scale;
    }
}

// ---------------------------------------------------------------------------
// Softmax over L (4096) per (batch, head) + weighted pool over head channels.
// pooled[n, h*64+d] = sum_l softmax(logits[n,:,h])[l] * X[n*4096+l, h*64+d]
// One block (256 threads) per (n,h).
// ---------------------------------------------------------------------------
__global__ __launch_bounds__(256) void softmax_pool(const float* __restrict__ logits, // [M,16]
                                                    const bf16_t* __restrict__ X,     // [M,1024]
                                                    float* __restrict__ pooled) {     // [8,1024]
    __shared__ float red[256];
    const int n = blockIdx.x >> 4;
    const int h = blockIdx.x & 15;
    const int tid = threadIdx.x;
    const float* lg = logits + (size_t)n * LSEQ * HEADS + h;

    // pass 1: max
    float m = -1e30f;
    for (int l = tid; l < LSEQ; l += 256) m = fmaxf(m, lg[(size_t)l * HEADS]);
    red[tid] = m;
    __syncthreads();
    for (int s = 128; s > 0; s >>= 1) {
        if (tid < s) red[tid] = fmaxf(red[tid], red[tid + s]);
        __syncthreads();
    }
    m = red[0];
    __syncthreads();

    // pass 2: sum of exp
    float ssum = 0.f;
    for (int l = tid; l < LSEQ; l += 256) ssum += __expf(lg[(size_t)l * HEADS] - m);
    red[tid] = ssum;
    __syncthreads();
    for (int s = 128; s > 0; s >>= 1) {
        if (tid < s) red[tid] += red[tid + s];
        __syncthreads();
    }
    const float inv = 1.0f / red[0];
    __syncthreads();

    // pass 3: weighted pool; 4 partial sums per channel d
    const int d = tid & 63;
    const int sub = tid >> 6;  // 0..3
    const bf16_t* xp = X + (size_t)n * LSEQ * CDIM + (size_t)h * 64 + d;
    float acc = 0.f;
    for (int l = sub; l < LSEQ; l += 4)
        acc += __expf(lg[(size_t)l * HEADS] - m) * (float)xp[(size_t)l * CDIM];
    red[tid] = acc;
    __syncthreads();
    if (sub == 0) {
        float t = red[tid] + red[tid + 64] + red[tid + 128] + red[tid + 192];
        pooled[(size_t)n * CDIM + h * 64 + d] = t * inv;
    }
}

// ---------------------------------------------------------------------------
// Host launch
// ---------------------------------------------------------------------------
extern "C" void kernel_launch(void* const* d_in, const int* in_sizes, int n_in,
                              void* d_out, int out_size, void* d_ws, size_t ws_size,
                              hipStream_t stream) {
    const float* x_q  = (const float*)d_in[0];
    const float* x_kv = (const float*)d_in[1];
    const float* Wq  = (const float*)d_in[2];
    const float* bq  = (const float*)d_in[3];
    const float* Wqa = (const float*)d_in[4];
    const float* bqa = (const float*)d_in[5];
    const float* Wk  = (const float*)d_in[6];
    const float* bk  = (const float*)d_in[7];
    const float* Wka = (const float*)d_in[8];
    const float* bka = (const float*)d_in[9];
    const float* Wv  = (const float*)d_in[10];
    const float* bv  = (const float*)d_in[11];
    const float* Wt  = (const float*)d_in[12];
    const float* bt  = (const float*)d_in[13];
    const float* Wp  = (const float*)d_in[14];
    const float* bp  = (const float*)d_in[15];
    float* out = (float*)d_out;

    // workspace carve-up (~269 MB)
    char* ws = (char*)d_ws;
    size_t off = 0;
    auto alloc = [&](size_t bytes) -> void* {
        void* p = ws + off;
        off += (bytes + 255) & ~(size_t)255;
        return p;
    };
    const size_t MK = (size_t)MROWS * CDIM;
    bf16_t* bufA = (bf16_t*)alloc(MK * 2);  // x_q bf16, later KV
    bf16_t* bufB = (bf16_t*)alloc(MK * 2);  // x_kv bf16, later H
    bf16_t* Qbf  = (bf16_t*)alloc(MK * 2);
    bf16_t* Kbf  = (bf16_t*)alloc(MK * 2);
    bf16_t* WqT  = (bf16_t*)alloc((size_t)CDIM * CDIM * 2);
    bf16_t* WkT  = (bf16_t*)alloc((size_t)CDIM * CDIM * 2);
    bf16_t* WvT  = (bf16_t*)alloc((size_t)CDIM * CDIM * 2);
    bf16_t* WtT  = (bf16_t*)alloc((size_t)CDIM * CDIM * 2);
    bf16_t* WpT  = (bf16_t*)alloc((size_t)CDIM * CDIM * 2);
    bf16_t* WqaT = (bf16_t*)alloc((size_t)HEADS * CDIM * 2);
    bf16_t* WkaT = (bf16_t*)alloc((size_t)HEADS * CDIM * 2);
    float*  logits  = (float*)alloc((size_t)MROWS * HEADS * 4);
    float*  pooledQ = (float*)alloc((size_t)8 * CDIM * 4);
    float*  pooledK = (float*)alloc((size_t)8 * CDIM * 4);
    (void)ws_size; (void)in_sizes; (void)n_in; (void)out_size;

    const dim3 gemmGrid(CDIM / 128, MROWS / 128);  // (8, 256)
    const dim3 blk256(256);
    const dim3 tBlock(32, 8);

    // Stage 0: conversions
    convert_bf16<<<2048, blk256, 0, stream>>>(x_q, bufA, MK);
    convert_bf16<<<2048, blk256, 0, stream>>>(x_kv, bufB, MK);
    transpose_convert<<<dim3(32, 32), tBlock, 0, stream>>>(Wq, WqT, CDIM, CDIM);
    transpose_convert<<<dim3(32, 32), tBlock, 0, stream>>>(Wk, WkT, CDIM, CDIM);
    transpose_convert<<<dim3(32, 32), tBlock, 0, stream>>>(Wv, WvT, CDIM, CDIM);
    transpose_convert<<<dim3(32, 32), tBlock, 0, stream>>>(Wt, WtT, CDIM, CDIM);
    transpose_convert<<<dim3(32, 32), tBlock, 0, stream>>>(Wp, WpT, CDIM, CDIM);
    transpose_convert<<<dim3(1, 32), tBlock, 0, stream>>>(Wqa, WqaT, CDIM, HEADS);
    transpose_convert<<<dim3(1, 32), tBlock, 0, stream>>>(Wka, WkaT, CDIM, HEADS);

    // Stage 1: queries = x_q @ Wq + bq (bf16)
    gemm_bf16<1><<<gemmGrid, blk256, 0, stream>>>(bufA, WqT, bq, nullptr, nullptr, Qbf);
    // Stage 2: q logits + softmax-pool -> pooled_q
    logits_kernel<<<256, blk256, 0, stream>>>(Qbf, nullptr, WqaT, bqa, logits);
    softmax_pool<<<128, blk256, 0, stream>>>(logits, Qbf, pooledQ);
    // Stage 3: keys = x_kv @ Wk + bk (bf16)
    gemm_bf16<1><<<gemmGrid, blk256, 0, stream>>>(bufB, WkT, bk, nullptr, nullptr, Kbf);
    // Stage 4: k logits (keys gated by pooled_q) + softmax-pool -> pooled_k
    logits_kernel<<<256, blk256, 0, stream>>>(Kbf, pooledQ, WkaT, bka, logits);
    softmax_pool<<<128, blk256, 0, stream>>>(logits, Kbf, pooledK);
    // Stage 5: KV = (x_kv @ Wv + bv) * pooled_k   (fused gating epilogue)
    gemm_bf16<2><<<gemmGrid, blk256, 0, stream>>>(bufB, WvT, bv, pooledK, nullptr, bufA);
    // Stage 6: H = KV @ Wt + bt + queries        (fused residual epilogue)
    gemm_bf16<3><<<gemmGrid, blk256, 0, stream>>>(bufA, WtT, bt, nullptr, Qbf, bufB);
    // Stage 7: out = H @ Wp + bp (f32)
    gemm_bf16<4><<<gemmGrid, blk256, 0, stream>>>(bufB, WpT, bp, nullptr, nullptr, out);
}
